// EncoderLayer_19516331393696
// MI455X (gfx1250) — matrix operand order repaired and checked by reading, not compiled, not run on hardware
//
#include <hip/hip_runtime.h>
#include <math.h>

// ---------------------------------------------------------------------------
// Shapes (fixed by the reference)
// ---------------------------------------------------------------------------
#define B_  4
#define S_  2048
#define D_  1024
#define H_  16
#define DK_ 64
#define DI_ 4096
#define M_  (B_ * S_)      // 8192 token rows
#define HV_ (H_ * DK_)     // 1024

// TDM-staged B panel: chunks of 256 bf16 along K, 64 rows (N), padded rows.
// Row = 128 dwords data + 4 dwords pad = 132 dwords -> bank stride 4 (conflict-free).
#define CHUNK_BF   256                 // bf16 elements per chunk row
#define LDS_ROW_BF 264                 // 132 dwords in bf16 elements

// Attention staged panels (per 32-column t-chunk):
//   K: 32 rows x 64 bf16, pad 4dw every 32dw  -> row stride 36 dw = 72 bf16
//   V: 64 rows x 32 bf16, pad 4dw every 16dw  -> row stride 20 dw = 40 bf16
#define KROW_BF 72
#define VROW_BF 40

typedef __attribute__((ext_vector_type(16))) __bf16 v16bf;
typedef __attribute__((ext_vector_type(8)))  __bf16 v8bf;
typedef __attribute__((ext_vector_type(8)))  float  v8f;
typedef __attribute__((ext_vector_type(4)))  unsigned int v4u;
typedef __attribute__((ext_vector_type(8)))  unsigned int v8u;

// ---------------------------------------------------------------------------
// WMMA helpers (CDNA5 bf16, 16x16x32, f32 accumulate)
// ---------------------------------------------------------------------------
__device__ inline v8f wmma_bf16(v16bf a, v16bf b, v8f c) {
    return __builtin_amdgcn_wmma_f32_16x16x32_bf16(
        false, a, false, b, (short)0, c, false, false);
}

// A operand: 16x32 bf16 tile, row-major. lane L -> row (L&15); hi=L>>4 selects
// ks {kb+hi*8..+7, kb+16+hi*8..+7}
__device__ inline v16bf load_a(const __bf16* __restrict__ base, long lda, int kbase, int lane) {
    int r = lane & 15, hi = lane >> 4;
    const __bf16* p = base + (long)r * lda + kbase + hi * 8;
    union { v16bf v; v8bf h[2]; } u;
    u.h[0] = *(const v8bf*)(p);
    u.h[1] = *(const v8bf*)(p + 16);
    return u.v;
}

// B operand from an LDS panel with padded rows: lane n holds row (row0+n),
// ks {kbase+hi*8..+7, kbase+16+hi*8..+7}.
__device__ inline v16bf load_bt_lds(const __bf16* panel, int row0, int rowstride,
                                    int kbase, int lane) {
    int n = lane & 15, hi = lane >> 4;
    const __bf16* p = panel + (long)(row0 + n) * rowstride + kbase + hi * 8;
    union { v16bf v; v8bf h[2]; } u;
    u.h[0] = *(const v8bf*)(p);
    u.h[1] = *(const v8bf*)(p + 16);
    return u.v;
}

// A operand assembled from an fp32 16x32 LDS tile (softmax probabilities).
__device__ inline v16bf load_a_lds_f32(const float* __restrict__ P, int ldp, int lane) {
    int r = lane & 15, hi = lane >> 4;
    const float* q = P + r * ldp + hi * 8;
    v16bf v;
#pragma unroll
    for (int i = 0; i < 8; ++i) {
        v[i]     = (__bf16)q[i];
        v[8 + i] = (__bf16)q[16 + i];
    }
    return v;
}

// ---------------------------------------------------------------------------
// Tensor Data Mover: 2D tile load global -> LDS with per-row padding.
//   tile_w_dw   : tile width in dwords
//   tile_rows   : number of rows
//   pad_code    : pad_interval code (3=16dw, 4=32dw, 6=128dw); pad_amount = 4 dwords
//   stride_dw   : tensor row stride in dwords
// ---------------------------------------------------------------------------
__device__ inline void tdm_load_2d(unsigned lds_off, const void* gptr,
                                   unsigned tile_w_dw, unsigned tile_rows,
                                   unsigned pad_code, unsigned stride_dw) {
    unsigned long long ga = (unsigned long long)(uintptr_t)gptr;
    v4u g0;
    g0[0] = 1u;                                           // count=1, user mode
    g0[1] = lds_off;                                      // lds_addr (bytes)
    g0[2] = (unsigned)(ga & 0xFFFFFFFFu);                 // global_addr[31:0]
    g0[3] = (unsigned)((ga >> 32) & 0x01FFFFFFu)          // global_addr[56:32]
          | 0x80000000u;                                  // type=2 (image)
    unsigned td0 = 1u << 20, td1 = 1u << 20;              // large tensor dims (no OOB clip)
    v8u g1;
    g1[0] = (2u << 16)            // data_size = 4B
          | (1u << 20)            // pad_enable
          | (pad_code << 22)      // pad_interval
          | (3u << 25);           // pad_amount: 4 dwords
    g1[1] = (td0 & 0xFFFFu) << 16;                        // tensor_dim0[15:0]
    g1[2] = (td0 >> 16) | ((td1 & 0xFFFFu) << 16);        // tensor_dim0[31:16], dim1[15:0]
    g1[3] = (td1 >> 16) | (tile_w_dw << 16);              // dim1[31:16], tile_dim0
    g1[4] = tile_rows & 0xFFFFu;                          // tile_dim1 (tile_dim2=0)
    g1[5] = stride_dw;                                    // tensor_dim0_stride[31:0]
    g1[6] = 0u;
    g1[7] = 0u;
    v4u z = {0u, 0u, 0u, 0u};
    asm volatile("tensor_load_to_lds %0, %1, %2, %3"
                 :: "s"(g0), "s"(g1), "s"(z), "s"(z)
                 : "memory");
}

// ---------------------------------------------------------------------------
// 32(M) x 64(N) wave GEMM body over one staged K-chunk.
// ---------------------------------------------------------------------------
__device__ inline void gemm_chunk_32x64(const __bf16* __restrict__ A, long lda, int kglobal,
                                        const __bf16* panel,
                                        v8f acc0[4], v8f acc1[4], int lane) {
#pragma unroll
    for (int k = 0; k < CHUNK_BF; k += 32) {
        v16bf a0 = load_a(A, lda, kglobal + k, lane);
        v16bf a1 = load_a(A + 16 * lda, lda, kglobal + k, lane);
#pragma unroll
        for (int c = 0; c < 4; ++c) {
            v16bf b = load_bt_lds(panel, c * 16, LDS_ROW_BF, k, lane);
            acc0[c] = wmma_bf16(a0, b, acc0[c]);
            acc1[c] = wmma_bf16(a1, b, acc1[c]);
        }
    }
}

// Double-buffered TDM pipeline over the full K extent: DMA of chunk i+1
// overlaps WMMA compute of chunk i.
__device__ inline void gemm_pipeline(const __bf16* __restrict__ A, long lda,
                                     const __bf16* __restrict__ Bt0, long ldb, int K,
                                     __bf16 (*panel)[64 * LDS_ROW_BF],
                                     v8f acc0[4], v8f acc1[4], int lane, int wid) {
    unsigned po[2] = { (unsigned)(uintptr_t)(&panel[0][0]),
                       (unsigned)(uintptr_t)(&panel[1][0]) };
    unsigned strdw = (unsigned)(ldb >> 1);
    int nch = K / CHUNK_BF;
    if (wid == 0) tdm_load_2d(po[0], Bt0, CHUNK_BF / 2, 64, 6, strdw);
    __builtin_amdgcn_s_wait_tensorcnt(0);
    __syncthreads();
    for (int i = 0; i < nch; ++i) {
        if (wid == 0 && i + 1 < nch)
            tdm_load_2d(po[(i + 1) & 1], Bt0 + (size_t)(i + 1) * CHUNK_BF,
                        CHUNK_BF / 2, 64, 6, strdw);
        gemm_chunk_32x64(A, lda, i * CHUNK_BF, panel[i & 1], acc0, acc1, lane);
        __builtin_amdgcn_s_wait_tensorcnt(0);
        __syncthreads();
    }
}

// ---------------------------------------------------------------------------
// Weight convert + transpose: src fp32 [R][C] -> dst bf16 [C][R]; batch via blockIdx.z
// ---------------------------------------------------------------------------
__global__ __launch_bounds__(256) void convT_kernel(const float* __restrict__ src,
                                                    __bf16* __restrict__ dst,
                                                    int R, int C) {
    size_t base = (size_t)blockIdx.z * R * C;
    long i = (long)blockIdx.x * 256 + threadIdx.x;
    if (i < (long)R * C) {
        int r = (int)(i / C), c = (int)(i % C);
        dst[base + (size_t)c * R + r] = (__bf16)src[base + i];
    }
}

// ---------------------------------------------------------------------------
// LayerNorm: fp32 row -> bf16 row
// ---------------------------------------------------------------------------
__global__ __launch_bounds__(256) void ln_kernel(const float* __restrict__ x,
                                                 const float* __restrict__ g,
                                                 const float* __restrict__ b,
                                                 __bf16* __restrict__ out) {
    __shared__ float r1[256], r2[256];
    int row = blockIdx.x, t = threadIdx.x;
    const float* xr = x + (size_t)row * D_;
    float v[4], s = 0.f, s2 = 0.f;
#pragma unroll
    for (int i = 0; i < 4; ++i) { float f = xr[t + i * 256]; v[i] = f; s += f; s2 += f * f; }
    r1[t] = s; r2[t] = s2; __syncthreads();
    for (int w = 128; w > 0; w >>= 1) { if (t < w) { r1[t] += r1[t + w]; r2[t] += r2[t + w]; } __syncthreads(); }
    float mean = r1[0] * (1.f / D_);
    float var  = r2[0] * (1.f / D_) - mean * mean;
    float inv  = rsqrtf(var + 1e-5f);
#pragma unroll
    for (int i = 0; i < 4; ++i) {
        int c = t + i * 256;
        out[(size_t)row * D_ + c] = (__bf16)((v[i] - mean) * inv * g[c] + b[c]);
    }
}

// Two chained LayerNorms (ln_ffn then ffn_ln) -> bf16
__global__ __launch_bounds__(256) void double_ln_kernel(const float* __restrict__ x,
                                                        const float* __restrict__ g1,
                                                        const float* __restrict__ b1,
                                                        const float* __restrict__ g2,
                                                        const float* __restrict__ b2,
                                                        __bf16* __restrict__ out) {
    __shared__ float r1[256], r2[256];
    int row = blockIdx.x, t = threadIdx.x;
    const float* xr = x + (size_t)row * D_;
    float v[4], s = 0.f, s2 = 0.f;
#pragma unroll
    for (int i = 0; i < 4; ++i) { float f = xr[t + i * 256]; v[i] = f; s += f; s2 += f * f; }
    r1[t] = s; r2[t] = s2; __syncthreads();
    for (int w = 128; w > 0; w >>= 1) { if (t < w) { r1[t] += r1[t + w]; r2[t] += r2[t + w]; } __syncthreads(); }
    float mean = r1[0] * (1.f / D_);
    float var  = r2[0] * (1.f / D_) - mean * mean;
    float inv  = rsqrtf(var + 1e-5f);
    __syncthreads();
    float rr[4]; s = 0.f; s2 = 0.f;
#pragma unroll
    for (int i = 0; i < 4; ++i) {
        int c = t + i * 256;
        float f = (v[i] - mean) * inv * g1[c] + b1[c];
        rr[i] = f; s += f; s2 += f * f;
    }
    r1[t] = s; r2[t] = s2; __syncthreads();
    for (int w = 128; w > 0; w >>= 1) { if (t < w) { r1[t] += r1[t + w]; r2[t] += r2[t + w]; } __syncthreads(); }
    mean = r1[0] * (1.f / D_);
    var  = r2[0] * (1.f / D_) - mean * mean;
    inv  = rsqrtf(var + 1e-5f);
#pragma unroll
    for (int i = 0; i < 4; ++i) {
        int c = t + i * 256;
        out[(size_t)row * D_ + c] = (__bf16)((rr[i] - mean) * inv * g2[c] + b2[c]);
    }
}

// ---------------------------------------------------------------------------
// QKV projection: h[M,D] @ {Wq,Wk,Wv}T[h][64][D]; V stored transposed [b,h,64,S]
// grid: (M/128, H*3); block 128 (4 waves x 32-row tiles); TDM double-buffered B
// ---------------------------------------------------------------------------
__global__ __launch_bounds__(128) void qkv_kernel(const __bf16* __restrict__ hbf,
                                                  const __bf16* __restrict__ WqT,
                                                  const __bf16* __restrict__ WkT,
                                                  const __bf16* __restrict__ WvT,
                                                  __bf16* __restrict__ Q,
                                                  __bf16* __restrict__ K,
                                                  __bf16* __restrict__ Vt) {
    __shared__ __bf16 panel[2][64 * LDS_ROW_BF];
    int lane = threadIdx.x & 31, wid = threadIdx.x >> 5;
    int m0 = blockIdx.x * 128 + wid * 32;
    int h  = blockIdx.y / 3;
    int which = blockIdx.y % 3;
    const __bf16* WT = (which == 0 ? WqT : which == 1 ? WkT : WvT) + (size_t)h * 64 * D_;
    const __bf16* A  = hbf + (size_t)m0 * D_;
    v8f acc0[4] = {}, acc1[4] = {};
    gemm_pipeline(A, D_, WT, D_, D_, panel, acc0, acc1, lane, wid);
    int n = lane & 15, hi = lane >> 4;
#pragma unroll
    for (int t = 0; t < 2; ++t) {
        v8f* acc = (t == 0) ? acc0 : acc1;
#pragma unroll
        for (int c = 0; c < 4; ++c)
#pragma unroll
            for (int j = 0; j < 8; ++j) {
                int m = m0 + t * 16 + 8 * hi + j;
                int b = m / S_, s = m % S_;
                int col = c * 16 + n;
                float val = acc[c][j];
                if (which == 0)
                    Q[(((size_t)b * H_ + h) * S_ + s) * 64 + col] = (__bf16)val;
                else if (which == 1)
                    K[(((size_t)b * H_ + h) * S_ + s) * 64 + col] = (__bf16)val;
                else
                    Vt[(((size_t)b * H_ + h) * 64 + col) * S_ + s] = (__bf16)val;
            }
    }
}

// ---------------------------------------------------------------------------
// Flash attention: 4 waves/block share one (b,h); K/V chunks staged by TDM
// (double-buffered); per-wave online softmax.
// ---------------------------------------------------------------------------
__global__ __launch_bounds__(128) void attn_kernel(const __bf16* __restrict__ Q,
                                                   const __bf16* __restrict__ K,
                                                   const __bf16* __restrict__ Vt,
                                                   const int* __restrict__ mask,
                                                   __bf16* __restrict__ o) {
    __shared__ __bf16 kpan[2][32 * KROW_BF];
    __shared__ __bf16 vpan[2][64 * VROW_BF];
    __shared__ float Plds[4][16 * 32];
    int lane = threadIdx.x & 31, wid = threadIdx.x >> 5;
    int tile = blockIdx.x * 4 + wid;
    int mt = tile & 127;          // S/16 = 128
    int hh = (tile >> 7) & 15;
    int b  = tile >> 11;
    int s0 = mt * 16;
    const __bf16* Qb = Q  + (((size_t)b * H_ + hh) * S_ + s0) * 64;
    const __bf16* Kb = K  + (((size_t)b * H_ + hh) * S_) * 64;
    const __bf16* Vb = Vt + (((size_t)b * H_ + hh) * 64) * S_;
    const int*    Mb = mask + ((size_t)b * S_ + s0) * S_;
    int n = lane & 15, hi = lane >> 4;

    unsigned ko[2] = { (unsigned)(uintptr_t)(&kpan[0][0]),
                       (unsigned)(uintptr_t)(&kpan[1][0]) };
    unsigned vo[2] = { (unsigned)(uintptr_t)(&vpan[0][0]),
                       (unsigned)(uintptr_t)(&vpan[1][0]) };

    v16bf qa0 = load_a(Qb, 64, 0, lane);
    v16bf qa1 = load_a(Qb, 64, 32, lane);

    v8f acc[4] = {};
    float rmax[8], rsum[8];
#pragma unroll
    for (int j = 0; j < 8; ++j) { rmax[j] = -3e38f; rsum[j] = 0.f; }
    float* P = &Plds[wid][0];

    if (wid == 0) {
        tdm_load_2d(ko[0], Kb, 32, 32, 4, 32);                      // K rows t0..t0+31
        tdm_load_2d(vo[0], Vb, 16, 64, 3, (unsigned)(S_ / 2));      // V cols t0..t0+31
    }
    __builtin_amdgcn_s_wait_tensorcnt(0);
    __syncthreads();

    for (int t0 = 0, i = 0; t0 < S_; t0 += 32, ++i) {
        if (wid == 0 && t0 + 32 < S_) {
            tdm_load_2d(ko[(i + 1) & 1], Kb + (size_t)(t0 + 32) * 64, 32, 32, 4, 32);
            tdm_load_2d(vo[(i + 1) & 1], Vb + (t0 + 32), 16, 64, 3, (unsigned)(S_ / 2));
        }
        const __bf16* kp = kpan[i & 1];
        const __bf16* vp = vpan[i & 1];

        v8f s1 = {}, s2 = {};
        s1 = wmma_bf16(qa0, load_bt_lds(kp, 0,  KROW_BF, 0,  lane), s1);
        s1 = wmma_bf16(qa1, load_bt_lds(kp, 0,  KROW_BF, 32, lane), s1);
        s2 = wmma_bf16(qa0, load_bt_lds(kp, 16, KROW_BF, 0,  lane), s2);
        s2 = wmma_bf16(qa1, load_bt_lds(kp, 16, KROW_BF, 32, lane), s2);

        float cmax[8];
#pragma unroll
        for (int j = 0; j < 8; ++j) {
            int m = 8 * hi + j;
            int mk1 = Mb[(size_t)m * S_ + t0 + n];
            int mk2 = Mb[(size_t)m * S_ + t0 + 16 + n];
            float a1 = mk1 ? s1[j] * 0.125f : -1e10f;
            float a2 = mk2 ? s2[j] * 0.125f : -1e10f;
            s1[j] = a1; s2[j] = a2;
            cmax[j] = fmaxf(a1, a2);
        }
#pragma unroll
        for (int off = 1; off <= 8; off <<= 1)
#pragma unroll
            for (int j = 0; j < 8; ++j)
                cmax[j] = fmaxf(cmax[j], __shfl_xor(cmax[j], off, 32));

        float csum[8], corr[8];
#pragma unroll
        for (int j = 0; j < 8; ++j) {
            float nm = fmaxf(rmax[j], cmax[j]);
            corr[j] = __expf(rmax[j] - nm);
            rmax[j] = nm;
            float p1 = __expf(s1[j] - nm);
            float p2 = __expf(s2[j] - nm);
            s1[j] = p1; s2[j] = p2;
            csum[j] = p1 + p2;
        }
#pragma unroll
        for (int off = 1; off <= 8; off <<= 1)
#pragma unroll
            for (int j = 0; j < 8; ++j)
                csum[j] += __shfl_xor(csum[j], off, 32);
#pragma unroll
        for (int j = 0; j < 8; ++j) rsum[j] = rsum[j] * corr[j] + csum[j];
#pragma unroll
        for (int c = 0; c < 4; ++c)
#pragma unroll
            for (int j = 0; j < 8; ++j) acc[c][j] *= corr[j];

        // C-layout -> A-layout transpose through per-wave LDS tile
#pragma unroll
        for (int j = 0; j < 8; ++j) {
            int m = 8 * hi + j;
            P[m * 32 + n]      = s1[j];
            P[m * 32 + 16 + n] = s2[j];
        }
        v16bf pa = load_a_lds_f32(P, 32, lane);
#pragma unroll
        for (int c = 0; c < 4; ++c) {
            v16bf vb = load_bt_lds(vp, c * 16, VROW_BF, 0, lane);
            acc[c] = wmma_bf16(pa, vb, acc[c]);
        }

        __builtin_amdgcn_s_wait_tensorcnt(0);
        __syncthreads();
    }

#pragma unroll
    for (int c = 0; c < 4; ++c)
#pragma unroll
        for (int j = 0; j < 8; ++j) {
            int m = 8 * hi + j;
            float val = acc[c][j] / rsum[j];
            o[((size_t)(b * S_ + s0 + m)) * HV_ + hh * 64 + c * 16 + n] = (__bf16)val;
        }
}

// ---------------------------------------------------------------------------
// Output projection + residual: attn = x + o @ Wo    (fp32 out)
// ---------------------------------------------------------------------------
__global__ __launch_bounds__(128) void outproj_kernel(const __bf16* __restrict__ o,
                                                      const __bf16* __restrict__ WoT,
                                                      const float* __restrict__ x,
                                                      float* __restrict__ attn) {
    __shared__ __bf16 panel[2][64 * LDS_ROW_BF];
    int lane = threadIdx.x & 31, wid = threadIdx.x >> 5;
    int m0 = blockIdx.x * 128 + wid * 32;
    int n0 = blockIdx.y * 64;
    v8f acc0[4] = {}, acc1[4] = {};
    gemm_pipeline(o + (size_t)m0 * HV_, HV_, WoT + (size_t)n0 * HV_, HV_, HV_,
                  panel, acc0, acc1, lane, wid);
    int n = lane & 15, hi = lane >> 4;
#pragma unroll
    for (int t = 0; t < 2; ++t) {
        v8f* acc = (t == 0) ? acc0 : acc1;
#pragma unroll
        for (int c = 0; c < 4; ++c)
#pragma unroll
            for (int j = 0; j < 8; ++j) {
                int m = m0 + t * 16 + 8 * hi + j, col = n0 + c * 16 + n;
                attn[(size_t)m * D_ + col] = x[(size_t)m * D_ + col] + acc[c][j];
            }
    }
}

// ---------------------------------------------------------------------------
// FFN1: g = gelu_erf(f @ W1 + b1)  -> bf16
// ---------------------------------------------------------------------------
__global__ __launch_bounds__(128) void ffn1_kernel(const __bf16* __restrict__ f,
                                                   const __bf16* __restrict__ W1T,
                                                   const float* __restrict__ b1,
                                                   __bf16* __restrict__ gbuf) {
    __shared__ __bf16 panel[2][64 * LDS_ROW_BF];
    int lane = threadIdx.x & 31, wid = threadIdx.x >> 5;
    int m0 = blockIdx.x * 128 + wid * 32;
    int n0 = blockIdx.y * 64;
    v8f acc0[4] = {}, acc1[4] = {};
    gemm_pipeline(f + (size_t)m0 * D_, D_, W1T + (size_t)n0 * D_, D_, D_,
                  panel, acc0, acc1, lane, wid);
    int n = lane & 15, hi = lane >> 4;
#pragma unroll
    for (int t = 0; t < 2; ++t) {
        v8f* acc = (t == 0) ? acc0 : acc1;
#pragma unroll
        for (int c = 0; c < 4; ++c)
#pragma unroll
            for (int j = 0; j < 8; ++j) {
                int m = m0 + t * 16 + 8 * hi + j, col = n0 + c * 16 + n;
                float v = acc[c][j] + b1[col];
                v = 0.5f * v * (1.f + erff(v * 0.70710678118f));
                gbuf[(size_t)m * DI_ + col] = (__bf16)v;
            }
    }
}

// ---------------------------------------------------------------------------
// FFN2 + residual: out = attn + g @ W2 + b2   (fp32 final output)
// ---------------------------------------------------------------------------
__global__ __launch_bounds__(128) void ffn2_kernel(const __bf16* __restrict__ gbuf,
                                                   const __bf16* __restrict__ W2T,
                                                   const float* __restrict__ b2,
                                                   const float* __restrict__ attn,
                                                   float* __restrict__ out) {
    __shared__ __bf16 panel[2][64 * LDS_ROW_BF];
    int lane = threadIdx.x & 31, wid = threadIdx.x >> 5;
    int m0 = blockIdx.x * 128 + wid * 32;
    int n0 = blockIdx.y * 64;
    v8f acc0[4] = {}, acc1[4] = {};
    gemm_pipeline(gbuf + (size_t)m0 * DI_, DI_, W2T + (size_t)n0 * DI_, DI_, DI_,
                  panel, acc0, acc1, lane, wid);
    int n = lane & 15, hi = lane >> 4;
#pragma unroll
    for (int t = 0; t < 2; ++t) {
        v8f* acc = (t == 0) ? acc0 : acc1;
#pragma unroll
        for (int c = 0; c < 4; ++c)
#pragma unroll
            for (int j = 0; j < 8; ++j) {
                int m = m0 + t * 16 + 8 * hi + j, col = n0 + c * 16 + n;
                out[(size_t)m * D_ + col] = attn[(size_t)m * D_ + col] + acc[c][j] + b2[col];
            }
    }
}

// ---------------------------------------------------------------------------
// Host-side orchestration
// ---------------------------------------------------------------------------
extern "C" void kernel_launch(void* const* d_in, const int* in_sizes, int n_in,
                              void* d_out, int out_size, void* d_ws, size_t ws_size,
                              hipStream_t stream) {
    const float* x    = (const float*)d_in[0];
    const int*   mask = (const int*)d_in[1];
    const float* Wq   = (const float*)d_in[2];
    const float* Wk   = (const float*)d_in[3];
    const float* Wv   = (const float*)d_in[4];
    const float* Wo   = (const float*)d_in[5];
    const float* lag  = (const float*)d_in[6];
    const float* lab  = (const float*)d_in[7];
    const float* lfg  = (const float*)d_in[8];
    const float* lfb  = (const float*)d_in[9];
    const float* fng  = (const float*)d_in[10];
    const float* fnb  = (const float*)d_in[11];
    const float* W1   = (const float*)d_in[12];
    const float* b1   = (const float*)d_in[13];
    const float* W2   = (const float*)d_in[14];
    const float* b2   = (const float*)d_in[15];
    float* out = (float*)d_out;

    char* ws = (char*)d_ws;
    size_t off = 0;
    auto alloc = [&](size_t bytes) -> void* {
        void* p = ws + off;
        off = (off + bytes + 255) & ~(size_t)255;
        return p;
    };

    __bf16* WqT  = (__bf16*)alloc((size_t)H_ * 64 * D_ * 2);
    __bf16* WkT  = (__bf16*)alloc((size_t)H_ * 64 * D_ * 2);
    __bf16* WvT  = (__bf16*)alloc((size_t)H_ * 64 * D_ * 2);
    __bf16* WoT  = (__bf16*)alloc((size_t)D_ * HV_ * 2);
    __bf16* W1T  = (__bf16*)alloc((size_t)DI_ * D_ * 2);
    __bf16* W2T  = (__bf16*)alloc((size_t)D_ * DI_ * 2);
    __bf16* hbf  = (__bf16*)alloc((size_t)M_ * D_ * 2);
    __bf16* Qb   = (__bf16*)alloc((size_t)B_ * H_ * S_ * 64 * 2);
    __bf16* Kb   = (__bf16*)alloc((size_t)B_ * H_ * S_ * 64 * 2);
    __bf16* Vtb  = (__bf16*)alloc((size_t)B_ * H_ * 64 * S_ * 2);
    __bf16* obf  = (__bf16*)alloc((size_t)M_ * HV_ * 2);
    float*  attn = (float*) alloc((size_t)M_ * D_ * 4);
    __bf16* fbf  = (__bf16*)alloc((size_t)M_ * D_ * 2);
    __bf16* gbf  = (__bf16*)alloc((size_t)M_ * DI_ * 2);
    (void)ws_size; (void)in_sizes; (void)n_in; (void)out_size;

    // 1) Weight convert + transpose to bf16 [N][K]
    convT_kernel<<<dim3((D_ * 64 + 255) / 256, 1, H_), 256, 0, stream>>>(Wq, WqT, D_, 64);
    convT_kernel<<<dim3((D_ * 64 + 255) / 256, 1, H_), 256, 0, stream>>>(Wk, WkT, D_, 64);
    convT_kernel<<<dim3((D_ * 64 + 255) / 256, 1, H_), 256, 0, stream>>>(Wv, WvT, D_, 64);
    convT_kernel<<<dim3((HV_ * D_ + 255) / 256, 1, 1), 256, 0, stream>>>(Wo, WoT, HV_, D_);
    convT_kernel<<<dim3((D_ * DI_ + 255) / 256, 1, 1), 256, 0, stream>>>(W1, W1T, D_, DI_);
    convT_kernel<<<dim3((DI_ * D_ + 255) / 256, 1, 1), 256, 0, stream>>>(W2, W2T, DI_, D_);

    // 2) Pre-attention LayerNorm -> bf16
    ln_kernel<<<M_, 256, 0, stream>>>(x, lag, lab, hbf);

    // 3) QKV projections (WMMA + double-buffered TDM weights), V stored transposed
    qkv_kernel<<<dim3(M_ / 128, H_ * 3), 128, 0, stream>>>(hbf, WqT, WkT, WvT, Qb, Kb, Vtb);

    // 4) Flash attention (TDM-staged K/V, WMMA QK^T + online softmax + WMMA PV)
    attn_kernel<<<dim3(B_ * H_ * (S_ / 16) / 4), 128, 0, stream>>>(Qb, Kb, Vtb, mask, obf);

    // 5) Output projection + residual
    outproj_kernel<<<dim3(M_ / 128, D_ / 64), 128, 0, stream>>>(obf, WoT, x, attn);

    // 6) Two chained LayerNorms -> bf16
    double_ln_kernel<<<M_, 256, 0, stream>>>(attn, lfg, lfb, fng, fnb, fbf);

    // 7) FFN
    ffn1_kernel<<<dim3(M_ / 128, DI_ / 64), 128, 0, stream>>>(fbf, W1T, b1, gbf);
    ffn2_kernel<<<dim3(M_ / 128, D_ / 64), 128, 0, stream>>>(gbf, W2T, b2, attn, out);
}
